// FiLM_23278722744464
// MI455X (gfx1250) — matrix operand order repaired and checked by reading, not compiled, not run on hardware
//
#include <hip/hip_runtime.h>
#include <hip/hip_bf16.h>
#include <math.h>

// ---------------- problem constants ----------------
#define SEQ   768
#define PRED  192
#define CIN   321
#define BATCHN 2
#define NORD  256
#define NMODE 32
#define BJ    (BATCHN*CIN)        // 642 rows
#define BJP   656                 // padded to 41*16
#define ACOLS (NMODE*2*NORD)      // 16384 (m, re/im, n)
#define LDA_PAD 8                 // bf16 elements of row padding in LDS (16B)

typedef __bf16 bf16;
typedef __attribute__((ext_vector_type(16))) __bf16 v16bf;
typedef __attribute__((ext_vector_type(8)))  float  v8f;

struct __attribute__((aligned(16))) U128 { unsigned int x, y, z, w; };
union FragU  { v16bf v; U128 q[2]; };
union FragHU { v16bf v; unsigned short s[16]; };

// ---- WMMA operand loaders (bf16, layouts per CDNA5 ISA 7.12.2) ----
// A-matrix 16x32: lane L (m = L&15, hi = L>>4) holds K = hi*8+[0..7] and 16+hi*8+[0..7]
__device__ inline v16bf load_fragA_bf16(const bf16* base, int row, int ld, int kc, int lane) {
  int hi = (lane >> 4) & 1;
  const bf16* p = base + (size_t)row * ld + kc + hi * 8;
  FragU f;
  f.q[0] = *(const U128*)(p);
  f.q[1] = *(const U128*)(p + 16);
  return f.v;
}
// B-matrix 32x16: lane L (n = L&15, hi = L>>4) holds K = hi*8+[0..7] and 16+hi*8+[0..7]
__device__ inline v16bf load_fragB_bf16(const bf16* base, int col, int ld, int kc, int lane) {
  int hi = (lane >> 4) & 1;
  const bf16* p = base + (size_t)col * ld + kc + hi * 8;
  FragU f;
  f.q[0] = *(const U128*)(p);
  f.q[1] = *(const U128*)(p + 16);
  return f.v;
}
// Same layouts but source data is f32 (convert on load)
__device__ inline v16bf load_fragA_f32(const float* base, int row, int ld, int kc, int lane) {
  int hi = (lane >> 4) & 1;
  const float* p = base + (size_t)row * ld + kc;
  v16bf f;
#pragma unroll
  for (int e = 0; e < 8; ++e) f[e] = (bf16)p[hi * 8 + e];
#pragma unroll
  for (int e = 0; e < 8; ++e) f[8 + e] = (bf16)p[16 + hi * 8 + e];
  return f;
}
__device__ inline v16bf load_fragB_f32(const float* base, int col, int ld, int kc, int lane) {
  int hi = (lane >> 4) & 1;
  const float* p = base + (size_t)col * ld + kc;
  v16bf f;
#pragma unroll
  for (int e = 0; e < 16; ++e) f[e] = (bf16)p[hi * 8 + (e & 7) + (e >> 3) * 16];
  return f;
}

// ---------------- Stage 0: per-row mean/std + normalized f (bf16) ----------------
__global__ void k_stats(const float* __restrict__ hist,
                        const float* __restrict__ aw, const float* __restrict__ ab,
                        float* __restrict__ meanB, float* __restrict__ stdB,
                        bf16* __restrict__ Fbf) {
  int bj  = blockIdx.x;            // 0..655
  int tid = threadIdx.x;           // 256 threads
  __shared__ float sh[256];
  if (bj >= BJ) {                  // zero pad rows
    if (tid == 0) { meanB[bj] = 0.f; stdB[bj] = 1.f; }
    for (int t = tid; t < SEQ; t += 256) Fbf[(size_t)bj * SEQ + t] = (bf16)0.f;
    return;
  }
  int b = bj / CIN, c = bj % CIN;
  const float* xp = hist + ((size_t)b * SEQ) * CIN + c;   // stride CIN over time
  float s = 0.f;
  for (int t = tid; t < SEQ; t += 256) s += xp[(size_t)t * CIN];
  sh[tid] = s; __syncthreads();
  for (int w = 128; w > 0; w >>= 1) { if (tid < w) sh[tid] += sh[tid + w]; __syncthreads(); }
  float mean = sh[0] * (1.f / SEQ);
  __syncthreads();
  float sq = 0.f;
  for (int t = tid; t < SEQ; t += 256) { float d = xp[(size_t)t * CIN] - mean; sq += d * d; }
  sh[tid] = sq; __syncthreads();
  for (int w = 128; w > 0; w >>= 1) { if (tid < w) sh[tid] += sh[tid + w]; __syncthreads(); }
  float stdev = sqrtf(sh[0] * (1.f / SEQ) + 1e-5f);
  if (tid == 0) { meanB[bj] = mean; stdB[bj] = stdev; }
  float w_ = aw[c], b_ = ab[c];
  for (int t = tid; t < SEQ; t += 256)
    Fbf[(size_t)bj * SEQ + t] = (bf16)((xp[(size_t)t * CIN] - mean) / stdev * w_ + b_);
}

__global__ void k_zero(float* p, int n) {
  int i = blockIdx.x * 256 + threadIdx.x;
  if (i < n) p[i] = 0.f;
}

// ---------------- Stage 1: data-independent precompute ----------------
__global__ void k_packA(const float* __restrict__ A, float* __restrict__ At) {
  int p = blockIdx.x, n = threadIdx.x;
  At[(size_t)p * NORD + n] = A[(size_t)n * NORD + p];
}
// v_k = A^k B, sequential chain (tiny: 0.2 GFLOP total). One block, 256 lanes.
__global__ void k_vk(const float* __restrict__ At, const float* __restrict__ Bv,
                     float* __restrict__ V, int Lw) {
  __shared__ float cur[NORD];
  int n = threadIdx.x;
  float c = Bv[n];
  cur[n] = c; V[n] = c;
  for (int k = 1; k < Lw; ++k) {
    __syncthreads();
    float acc = 0.f;
#pragma unroll 4
    for (int p = 0; p < NORD; ++p) acc += At[(size_t)p * NORD + n] * cur[p];
    __syncthreads();
    cur[n] = acc;
    V[(size_t)k * NORD + n] = acc;
  }
}
// T[s,n,m] = w_m^s * sum_{k<=Lw-1-s} v_k[n] w_m^k ; stored K-major (s contiguous)
__global__ void k_buildT(const float* __restrict__ V, bf16* __restrict__ Tt, int Lw) {
  int n = threadIdx.x;      // 0..255
  int m = blockIdx.x;       // 0..31
  float w = -6.283185307179586f * (float)m / (float)Lw;
  bf16* Tre = Tt + ((size_t)(2 * m)     * NORD + n) * Lw;
  bf16* Tim = Tt + ((size_t)(2 * m + 1) * NORD + n) * Lw;
  float ure = 0.f, uim = 0.f;
  for (int k = 0; k < Lw; ++k) {
    float vk = V[(size_t)k * NORD + n];
    float sk, ck; __sincosf(w * (float)k, &sk, &ck);
    ure += vk * ck; uim += vk * sk;
    int s = Lw - 1 - k;
    float ss, cs; __sincosf(w * (float)s, &ss, &cs);
    Tre[s] = (bf16)(ure * cs - uim * ss);
    Tim[s] = (bf16)(ure * ss + uim * cs);
  }
}
// Pack W transposed for B-operand: Wp[((2m+ri)*256 + o)*256 + i] = (wr|wi)[scale,i,o,m]
__global__ void k_packW(const float* __restrict__ wr, const float* __restrict__ wi,
                        bf16* __restrict__ Wp, int scale) {
  size_t tid = (size_t)blockIdx.x * 256 + threadIdx.x;  // 32*2*256*256 total
  int i  = (int)(tid & 255);
  int o  = (int)((tid >> 8) & 255);
  int rm = (int)(tid >> 16);          // 2m+ri
  int ri = rm & 1, m = rm >> 1;
  const float* src = ri ? wi : wr;
  float val = src[(((size_t)scale * NORD + i) * NORD + o) * NMODE + m];
  Wp[((size_t)rm * NORD + o) * NORD + i] = (bf16)val;
}

// ---------------- Stage 2: a = f @ T  (656 x 16384 x Lw) ----------------
// Block = 8 waves. The 16-row A panel (16 x Lw bf16, <=24KB) is staged into LDS
// once per block with ASYNC global->LDS copies (ASYNCcnt / s_wait_asynccnt),
// then every wave builds A fragments with ds_load_b128 and computes a 16x32
// output slab (2 accumulators per A fragment).
__global__ void __launch_bounds__(256) k_gemm1(const bf16* __restrict__ F,
                                               const bf16* __restrict__ Tt,
                                               bf16* __restrict__ Abf, int Lw) {
  __shared__ bf16 sA[16 * (768 + LDA_PAD)];
  int tid  = threadIdx.x;
  int lane = tid & 31;
  int wave = tid >> 5;                 // 0..7
  int rowBase = blockIdx.x * 16;       // 41 tiles over bj
  int ldl = Lw + LDA_PAD;

  // ---- async stage: 16 rows x Lw bf16 from F (suffix window) into LDS ----
  {
    int cpr = Lw >> 3;                 // 16B chunks per row
    int nchunk = 16 * cpr;
    for (int ch = tid; ch < nchunk; ch += 256) {
      int r  = ch / cpr;
      int kc = (ch - r * cpr) * 8;
      const bf16* gp = F + (size_t)(rowBase + r) * SEQ + (SEQ - Lw) + kc;
      unsigned int lds_off = (unsigned int)(uintptr_t)(&sA[r * ldl + kc]);
      unsigned long long ga = (unsigned long long)(uintptr_t)gp;
      asm volatile("global_load_async_to_lds_b128 %0, %1, off"
                   :: "v"(lds_off), "v"(ga) : "memory");
    }
    asm volatile("s_wait_asynccnt 0" ::: "memory");
    __syncthreads();
  }

  // ---- WMMA loop: each wave owns two adjacent 16-col tiles ----
  int colBase = (blockIdx.y * 16 + wave * 2) * 16;   // 64 y-blocks * 256 cols
  int hi = lane >> 4, n16 = lane & 15;
  const bf16* B0 = Tt + (size_t)(colBase + n16) * Lw;
  const bf16* B1 = B0 + (size_t)16 * Lw;
  const bf16* pa_row = &sA[n16 * ldl];
  v8f acc0 = {0.f, 0.f, 0.f, 0.f, 0.f, 0.f, 0.f, 0.f};
  v8f acc1 = {0.f, 0.f, 0.f, 0.f, 0.f, 0.f, 0.f, 0.f};
  for (int kc = 0; kc < Lw; kc += 32) {
    FragU fa;
    const bf16* pa = pa_row + kc + hi * 8;
    fa.q[0] = *(const U128*)(pa);
    fa.q[1] = *(const U128*)(pa + 16);
    FragU fb0, fb1;
    const bf16* pb0 = B0 + kc + hi * 8;
    const bf16* pb1 = B1 + kc + hi * 8;
    fb0.q[0] = *(const U128*)(pb0); fb0.q[1] = *(const U128*)(pb0 + 16);
    fb1.q[0] = *(const U128*)(pb1); fb1.q[1] = *(const U128*)(pb1 + 16);
    acc0 = __builtin_amdgcn_wmma_f32_16x16x32_bf16(false, fa.v, false, fb0.v,
                                                   (short)0, acc0, false, false);
    acc1 = __builtin_amdgcn_wmma_f32_16x16x32_bf16(false, fa.v, false, fb1.v,
                                                   (short)0, acc1, false, false);
  }
#pragma unroll
  for (int r = 0; r < 8; ++r) {
    size_t rowi = (size_t)(rowBase + r + 8 * hi) * ACOLS;
    Abf[rowi + colBase + n16]      = (bf16)acc0[r];
    Abf[rowi + colBase + 16 + n16] = (bf16)acc1[r];
  }
}

// ---------------- Stage 3: complex mode mix + irfft-at-t0 fold ----------------
// x_dec_c[bj,o] = sum_m Re(gamma_m * sum_i a[bj,i,m] W[i,o,m])
__global__ void __launch_bounds__(128) k_gemm2(const bf16* __restrict__ Abf,
                                               const bf16* __restrict__ Wp,
                                               float* __restrict__ Xc, int Lw) {
  int lane = threadIdx.x & 31;
  int wave = threadIdx.x >> 5;          // 0..3, splits modes
  int rowBase = blockIdx.x * 16;        // 41 tiles over bj
  int colBase = blockIdx.y * 16;        // 16 tiles over o
  int row = rowBase + (lane & 15);
  int col = colBase + (lane & 15);
  v8f xacc = {0.f, 0.f, 0.f, 0.f, 0.f, 0.f, 0.f, 0.f};
  for (int m = wave; m < NMODE; m += 4) {
    const bf16* Are = Abf + (size_t)(2 * m)     * NORD;
    const bf16* Aim = Abf + (size_t)(2 * m + 1) * NORD;
    const bf16* Wre = Wp  + (size_t)(2 * m)     * NORD * NORD;
    const bf16* Wim = Wp  + (size_t)(2 * m + 1) * NORD * NORD;
    v8f cre = {0.f, 0.f, 0.f, 0.f, 0.f, 0.f, 0.f, 0.f};
    v8f cim = {0.f, 0.f, 0.f, 0.f, 0.f, 0.f, 0.f, 0.f};
    for (int kc = 0; kc < NORD; kc += 32) {
      v16bf ar = load_fragA_bf16(Are, row, ACOLS, kc, lane);
      v16bf ai = load_fragA_bf16(Aim, row, ACOLS, kc, lane);
      v16bf wr_ = load_fragB_bf16(Wre, col, NORD, kc, lane);
      v16bf wi_ = load_fragB_bf16(Wim, col, NORD, kc, lane);
      FragHU u; u.v = ai;                       // -a_im (flip sign bits)
#pragma unroll
      for (int e = 0; e < 16; ++e) u.s[e] ^= 0x8000u;
      cre = __builtin_amdgcn_wmma_f32_16x16x32_bf16(false, ar,  false, wr_, (short)0, cre, false, false);
      cre = __builtin_amdgcn_wmma_f32_16x16x32_bf16(false, u.v, false, wi_, (short)0, cre, false, false);
      cim = __builtin_amdgcn_wmma_f32_16x16x32_bf16(false, ar,  false, wi_, (short)0, cim, false, false);
      cim = __builtin_amdgcn_wmma_f32_16x16x32_bf16(false, ai,  false, wr_, (short)0, cim, false, false);
    }
    float ang = 6.283185307179586f * (float)m * (float)(PRED - 1) / (float)Lw;
    float wm  = (m == 0) ? 1.0f : 2.0f;
    float gre = wm * cosf(ang) / (float)Lw;
    float gim = wm * sinf(ang) / (float)Lw;
#pragma unroll
    for (int r = 0; r < 8; ++r) xacc[r] += gre * cre[r] - gim * cim[r];
  }
  __shared__ float red[4][32][8];
#pragma unroll
  for (int r = 0; r < 8; ++r) red[wave][lane][r] = xacc[r];
  __syncthreads();
  if (wave == 0) {
    int hi = lane >> 4, n = lane & 15;
#pragma unroll
    for (int r = 0; r < 8; ++r) {
      float s = red[0][lane][r] + red[1][lane][r] + red[2][lane][r] + red[3][lane][r];
      Xc[(size_t)(rowBase + r + 8 * hi) * NORD + colBase + n] = s;
    }
  }
}

// ---------------- Stage 4: Legendre eval projection + scale combine ----------------
// OutAcc[bj,p] += mlp_w[scale] * sum_o Xc[bj,o] * E[off+p, o]
__global__ void __launch_bounds__(32) k_xdecE(const float* __restrict__ Xc,
                                              const float* __restrict__ E,
                                              const float* __restrict__ mlpw,
                                              float* __restrict__ OutAcc,
                                              int scale, int Erows) {
  int lane = threadIdx.x;
  int rowBase = blockIdx.x * 16;     // 41 tiles over bj
  int colBase = blockIdx.y * 16;     // 12 tiles over p
  int row = rowBase + (lane & 15);
  int pg  = (Erows - PRED) + colBase + (lane & 15);   // global E row for this column
  v8f acc = {0.f, 0.f, 0.f, 0.f, 0.f, 0.f, 0.f, 0.f};
  for (int kc = 0; kc < NORD; kc += 32) {
    v16bf af = load_fragA_f32(Xc, row, NORD, kc, lane);
    v16bf bf_ = load_fragB_f32(E, pg, NORD, kc, lane);
    acc = __builtin_amdgcn_wmma_f32_16x16x32_bf16(false, af, false, bf_,
                                                  (short)0, acc, false, false);
  }
  float wk = mlpw[scale];
  int hi = lane >> 4, n = lane & 15;
#pragma unroll
  for (int r = 0; r < 8; ++r) {
    size_t idx = (size_t)(rowBase + r + 8 * hi) * PRED + colBase + n;
    OutAcc[idx] += wk * acc[r];
  }
}

// ---------------- Stage 5: de-normalize + transpose to (b, p, c) ----------------
__global__ void k_final(const float* __restrict__ OutAcc,
                        const float* __restrict__ meanB, const float* __restrict__ stdB,
                        const float* __restrict__ aw, const float* __restrict__ ab,
                        const float* __restrict__ mlpb, float* __restrict__ out) {
  int idx = blockIdx.x * 256 + threadIdx.x;
  if (idx >= BATCHN * PRED * CIN) return;
  int c = idx % CIN;
  int p = (idx / CIN) % PRED;
  int b = idx / (CIN * PRED);
  int bj = b * CIN + c;
  float v = OutAcc[(size_t)bj * PRED + p] + mlpb[0];
  v = (v - ab[c]) / (aw[c] + 1e-10f);
  v = v * stdB[bj] + meanB[bj];
  out[idx] = v;
}

// ---------------- host ----------------
extern "C" void kernel_launch(void* const* d_in, const int* in_sizes, int n_in,
                              void* d_out, int out_size, void* d_ws, size_t ws_size,
                              hipStream_t stream) {
  const float* hist  = (const float*)d_in[0];
  const float* aw    = (const float*)d_in[2];
  const float* ab    = (const float*)d_in[3];
  const float* Astk  = (const float*)d_in[4];   // (3,256,256)
  const float* Bstk  = (const float*)d_in[5];   // (3,256)
  const float* Earr[3] = {(const float*)d_in[6], (const float*)d_in[7], (const float*)d_in[8]};
  const float* wr    = (const float*)d_in[9];   // (3,256,256,32)
  const float* wi    = (const float*)d_in[10];
  const float* mlpw  = (const float*)d_in[11];  // (1,3)
  const float* mlpb  = (const float*)d_in[12];  // (1,)
  float* out = (float*)d_out;

  char* ws = (char*)d_ws;
  size_t cur = 0;
  auto take = [&](size_t bytes) { void* p = ws + cur; cur = (cur + bytes + 255) & ~(size_t)255; return p; };
  float* meanB  = (float*)take(BJP * 4);
  float* stdB   = (float*)take(BJP * 4);
  bf16*  Fbf    = (bf16*) take((size_t)BJP * SEQ * 2);
  float* At     = (float*)take((size_t)NORD * NORD * 4);
  float* V      = (float*)take((size_t)768 * NORD * 4);
  bf16*  Wp     = (bf16*) take((size_t)NMODE * 2 * NORD * NORD * 2);
  bf16*  Tt     = (bf16*) take((size_t)ACOLS * 768 * 2);
  bf16*  Abf    = (bf16*) take((size_t)BJP * ACOLS * 2);
  float* Xc     = (float*)take((size_t)BJP * NORD * 4);
  float* OutAcc = (float*)take((size_t)BJP * PRED * 4);
  (void)ws_size; (void)in_sizes; (void)n_in; (void)out_size;

  k_stats<<<BJP, 256, 0, stream>>>(hist, aw, ab, meanB, stdB, Fbf);
  k_zero<<<(BJP * PRED + 255) / 256, 256, 0, stream>>>(OutAcc, BJP * PRED);

  for (int s = 0; s < 3; ++s) {
    int ms = 1 << s;
    int Lw = PRED * ms;
    k_packA <<<NORD, 256, 0, stream>>>(Astk + (size_t)s * NORD * NORD, At);
    k_vk    <<<1, 256, 0, stream>>>(At, Bstk + (size_t)s * NORD, V, Lw);
    k_buildT<<<NMODE, 256, 0, stream>>>(V, Tt, Lw);
    k_packW <<<(NMODE * 2 * NORD * NORD) / 256, 256, 0, stream>>>(wr, wi, Wp, s);
    k_gemm1 <<<dim3(BJP / 16, ACOLS / 256), 256, 0, stream>>>(Fbf, Tt, Abf, Lw);
    k_gemm2 <<<dim3(BJP / 16, NORD / 16), 128, 0, stream>>>(Abf, Wp, Xc, Lw);
    k_xdecE <<<dim3(BJP / 16, PRED / 16), 32, 0, stream>>>(Xc, Earr[s], mlpw, OutAcc, s, PRED * ms);
  }
  k_final<<<(BATCHN * PRED * CIN + 255) / 256, 256, 0, stream>>>(OutAcc, meanB, stdB, aw, ab, mlpb, out);
}
